// EncoderDecoder_with_Attention_78142634983605
// MI455X (gfx1250) — matrix operand order repaired and checked by reading, not compiled, not run on hardware
//
#include <hip/hip_runtime.h>
#include <hip/hip_bf16.h>
#include <cstdint>

// ---------------------------------------------------------------------------
// MI455X (gfx1250) implementation notes:
//  * wave32; WMMA 16x16x32 bf16 -> f32 for all GEMMs
//  * all large streamed operands converted to bf16.
//  * L2 policy: W_out bf16 (98MB) + graph tensors (34MB) are kept cache-
//    friendly (RT loads, reused all 64 steps); the sentence-side streams
//    (ha_s + sentence bf16 = 134MB/step, no intra-step reuse) are loaded
//    non-temporally so they don't evict the resident set. HBM roofline then
//    ~134MB/step -> ~0.37ms total at 23.3 TB/s.
//  * Projection GEMM: each wave computes BOTH m-tiles (M=32) per n-tile,
//    loading each W_out B-fragment once -> 2 WMMAs per B fetch.
// ---------------------------------------------------------------------------

typedef unsigned short u16;
typedef __attribute__((ext_vector_type(8)))  u16    u16x8;
typedef __attribute__((ext_vector_type(16))) u16    u16x16;
typedef __attribute__((ext_vector_type(16))) __bf16 v16bf;
typedef __attribute__((ext_vector_type(8)))  float  v8f;

__device__ __forceinline__ float bf2f(u16 u) {
  return __builtin_bit_cast(float, ((unsigned)u) << 16);
}
__device__ __forceinline__ u16 f2bf(float f) {
  unsigned u = __builtin_bit_cast(unsigned, f);
  u += 0x7FFFu + ((u >> 16) & 1u);           // round-to-nearest-even
  return (u16)(u >> 16);
}

// ---------------------------------------------------------------------------
// f32 -> bf16 elementwise
__global__ void to_bf16_kernel(const float* __restrict__ src,
                               u16* __restrict__ dst, size_t n) {
  size_t i = (size_t)blockIdx.x * blockDim.x + threadIdx.x;
  if (i < n) dst[i] = f2bf(src[i]);
}

// dst[c*R + r] = bf16(src[r*C + c])   (transpose W_a so WMMA B-loads are
// contiguous along K)
__global__ void transpose_bf16_kernel(const float* __restrict__ src,
                                      u16* __restrict__ dst, int R, int C) {
  int i = blockIdx.x * blockDim.x + threadIdx.x;
  if (i < R * C) {
    int r = i / C, c = i % C;
    dst[(size_t)c * R + r] = f2bf(src[i]);
  }
}

__global__ void fill_int_kernel(int* p, int v, int n) {
  int i = blockIdx.x * blockDim.x + threadIdx.x;
  if (i < n) p[i] = v;
}

// ---------------------------------------------------------------------------
// C[M,N] = A[M,K](bf16 row-major) * W[N,K](bf16 row-major)^T  (+bias)
// One wave computes MREP vertically-stacked 16x16 C tiles for one n-tile,
// so each B fragment is fetched once and fed to MREP WMMAs.
// VGPR layouts used here (A/B k order is this code's contiguous choice, same for both):
//   A (16x32 bf16): lane L holds row M=L%16; half e -> K = e + 16*(L/16)
//   B (32x16 bf16): lane L holds col N=L%16; half e -> K = e + 16*(L/16)
//   C (16x16 f32):  VGPR r, lane L -> M = r + 8*(L/16), N = L%16
template <bool WITH_BIAS, bool OUT_F32, int MREP>
__global__ __launch_bounds__(256)
void wmma_gemm_bt(const u16* __restrict__ A, const u16* __restrict__ W,
                  const float* __restrict__ bias, void* __restrict__ out,
                  int M, int N, int K, int ldc) {
  int wave = (blockIdx.x * blockDim.x + threadIdx.x) >> 5;
  int lane = threadIdx.x & 31;
  int mGroups = (M >> 4) / MREP, nTiles = N >> 4;
  if (wave >= mGroups * nTiles) return;
  int mg = wave % mGroups, nt = wave / mGroups;
  int m0 = mg * (MREP << 4), n0 = nt << 4;
  int lm = lane & 15;
  int hi = lane >> 4;  // 0 or 1

  const u16* arow[MREP];
#pragma unroll
  for (int r = 0; r < MREP; ++r)
    arow[r] = A + (size_t)(m0 + (r << 4) + lm) * K + (hi ? 16 : 0);
  const u16* wrow = W + (size_t)(n0 + lm) * K + (hi ? 16 : 0);

  v8f c[MREP];
#pragma unroll
  for (int r = 0; r < MREP; ++r) c[r] = (v8f){};

  for (int k = 0; k < K; k += 32) {
    u16x16 bu = *(const u16x16*)(wrow + k);
    v16bf b = __builtin_bit_cast(v16bf, bu);
#pragma unroll
    for (int r = 0; r < MREP; ++r) {
      u16x8 alo = *(const u16x8*)(arow[r] + k);
      u16x8 ahi = *(const u16x8*)(arow[r] + k + 8);
      u16x16 au;
#pragma unroll
      for (int i = 0; i < 8; ++i) { au[i] = alo[i]; au[8 + i] = ahi[i]; }
      v16bf a = __builtin_bit_cast(v16bf, au);
      c[r] = __builtin_amdgcn_wmma_f32_16x16x32_bf16(false, a, false, b,
                                                     (short)0, c[r], false, false);
    }
  }

  float bv = WITH_BIAS ? bias[n0 + lm] : 0.0f;
#pragma unroll
  for (int r = 0; r < MREP; ++r) {
    int rbase = m0 + (r << 4) + (hi ? 8 : 0);
    if (OUT_F32) {
      float* o = (float*)out;
#pragma unroll
      for (int q = 0; q < 8; ++q)
        o[(size_t)(rbase + q) * ldc + n0 + lm] = c[r][q] + bv;
    } else {
      u16* o = (u16*)out;
#pragma unroll
      for (int q = 0; q < 8; ++q)
        o[(size_t)(rbase + q) * ldc + n0 + lm] = f2bf(c[r][q] + bv);
    }
  }
}

// ---------------------------------------------------------------------------
// score[b,k] = dot(ha_bf16[b,k,:H], h[b,:H]);  one wave per (b,k), H=512
// NT = non-temporal loads for the streamed ha tensor (no intra-step reuse).
template <bool NT>
__global__ __launch_bounds__(256)
void attn_score_kernel(const u16* __restrict__ ha, const float* __restrict__ hvec,
                       float* __restrict__ score, int Kdim) {
  __shared__ float hl[512];
  int b = blockIdx.y, tid = threadIdx.x;
  for (int i = tid; i < 512; i += 256) hl[i] = hvec[(size_t)b * 512 + i];
  __syncthreads();
  int wave = tid >> 5, lane = tid & 31;
  int k = blockIdx.x * 8 + wave;
  if (k >= Kdim) return;
  const u16* row = ha + ((size_t)b * Kdim + k) * 512 + lane * 16;
  u16x8 p0, p1;
  if (NT) {
    p0 = __builtin_nontemporal_load((const u16x8*)(row));
    p1 = __builtin_nontemporal_load((const u16x8*)(row + 8));
  } else {
    p0 = *(const u16x8*)(row);
    p1 = *(const u16x8*)(row + 8);
  }
  float s = 0.f;
#pragma unroll
  for (int i = 0; i < 8; ++i) {
    s += bf2f(p0[i]) * hl[lane * 16 + i];
    s += bf2f(p1[i]) * hl[lane * 16 + 8 + i];
  }
  for (int off = 16; off; off >>= 1) s += __shfl_xor(s, off, 32);
  if (lane == 0) score[(size_t)b * Kdim + k] = s;
}

// in-place softmax over a row of length Kdim (one block per b)
__global__ __launch_bounds__(256)
void softmax_row_kernel(float* __restrict__ s, int Kdim) {
  __shared__ float red[256];
  int b = blockIdx.x, tid = threadIdx.x;
  float* row = s + (size_t)b * Kdim;
  float m = -3.4e38f;
  for (int k = tid; k < Kdim; k += 256) m = fmaxf(m, row[k]);
  red[tid] = m; __syncthreads();
  for (int o = 128; o; o >>= 1) {
    if (tid < o) red[tid] = fmaxf(red[tid], red[tid + o]);
    __syncthreads();
  }
  m = red[0]; __syncthreads();
  float acc = 0.f;
  for (int k = tid; k < Kdim; k += 256) {
    float e = expf(row[k] - m); row[k] = e; acc += e;
  }
  red[tid] = acc; __syncthreads();
  for (int o = 128; o; o >>= 1) {
    if (tid < o) red[tid] += red[tid + o];
    __syncthreads();
  }
  float inv = 1.0f / red[0];
  for (int k = tid; k < Kdim; k += 256) row[k] *= inv;
}

// ct[b,h] = sum_k at[b,k] * enc_bf16[b,k,h] ; writes bf16 into X[b, dstCol+h]
template <bool NT>
__global__ __launch_bounds__(256)
void attn_context_kernel(const u16* __restrict__ enc, const float* __restrict__ at,
                         u16* __restrict__ X, int Kdim, int dstCol) {
  __shared__ float al[256];
  int b = blockIdx.y;
  int h = blockIdx.x * 256 + threadIdx.x;   // H = 512, grid.x = 2
  float acc = 0.f;
  for (int k0 = 0; k0 < Kdim; k0 += 256) {
    __syncthreads();
    al[threadIdx.x] = at[(size_t)b * Kdim + k0 + threadIdx.x];
    __syncthreads();
    const u16* base = enc + ((size_t)b * Kdim + k0) * 512 + h;
    for (int kk = 0; kk < 256; ++kk) {
      u16 u = NT ? __builtin_nontemporal_load(base + (size_t)kk * 512)
                 : base[(size_t)kk * 512];
      acc += al[kk] * bf2f(u);
    }
  }
  X[(size_t)b * 1536 + dstCol + h] = f2bf(acc);
}

// GRU cell: thread (b=blockIdx.x, j=threadIdx.x) over H=512
__global__ __launch_bounds__(512)
void gru_step_kernel(const float* __restrict__ emb_table, const int* __restrict__ inp,
                     const float* __restrict__ hcur,
                     const float* __restrict__ W_ih, const float* __restrict__ W_hh,
                     const float* __restrict__ b_ih, const float* __restrict__ b_hh,
                     float* __restrict__ hnext, u16* __restrict__ X) {
  __shared__ float e[512], hh[512];
  int b = blockIdx.x, j = threadIdx.x;
  int idx = inp[b];
  e[j]  = emb_table[(size_t)idx * 512 + j];
  hh[j] = hcur[(size_t)b * 512 + j];
  __syncthreads();
  float xr = b_ih[j], xz = b_ih[512 + j], xn = b_ih[1024 + j];
  float hr = b_hh[j], hz = b_hh[512 + j], hn = b_hh[1024 + j];
  const float* wr = W_ih + (size_t)j * 512;
  const float* wz = W_ih + (size_t)(512 + j) * 512;
  const float* wn = W_ih + (size_t)(1024 + j) * 512;
  const float* vr = W_hh + (size_t)j * 512;
  const float* vz = W_hh + (size_t)(512 + j) * 512;
  const float* vn = W_hh + (size_t)(1024 + j) * 512;
  for (int i = 0; i < 512; ++i) {
    float ei = e[i], hi = hh[i];
    xr += ei * wr[i]; xz += ei * wz[i]; xn += ei * wn[i];
    hr += hi * vr[i]; hz += hi * vz[i]; hn += hi * vn[i];
  }
  float r = 1.f / (1.f + expf(-(xr + hr)));
  float z = 1.f / (1.f + expf(-(xz + hz)));
  float n = tanhf(xn + r * hn);
  float hN = (1.f - z) * n + z * hh[j];
  hnext[(size_t)b * 512 + j] = hN;
  X[(size_t)b * 1536 + j] = f2bf(hN);
}

// first-occurrence argmax over V per batch row (matches jnp.argmax)
__global__ __launch_bounds__(256)
void argmax_row_kernel(const float* __restrict__ out, int* __restrict__ inp, int Vn) {
  __shared__ float bv[256];
  __shared__ int   bi[256];
  int b = blockIdx.x, tid = threadIdx.x;
  const float* row = out + (size_t)b * Vn;
  float best = -3.4e38f; int bidx = 0x7FFFFFFF;
  for (int v = tid; v < Vn; v += 256) {
    float x = row[v];
    if (x > best) { best = x; bidx = v; }
  }
  bv[tid] = best; bi[tid] = bidx; __syncthreads();
  for (int o = 128; o; o >>= 1) {
    if (tid < o) {
      if (bv[tid + o] > bv[tid] ||
          (bv[tid + o] == bv[tid] && bi[tid + o] < bi[tid])) {
        bv[tid] = bv[tid + o]; bi[tid] = bi[tid + o];
      }
    }
    __syncthreads();
  }
  if (tid == 0) inp[b] = bi[0];
}

// ---------------------------------------------------------------------------
extern "C" void kernel_launch(void* const* d_in, const int* in_sizes, int n_in,
                              void* d_out, int out_size, void* d_ws, size_t ws_size,
                              hipStream_t stream) {
  (void)in_sizes; (void)n_in; (void)out_size; (void)ws_size;
  const int B = 32, KS = 2048, KG = 512, H = 512, V = 32000, T = 64, H3 = 1536;

  const float* sent   = (const float*)d_in[0];   // (B,KS,H)
  const float* graph  = (const float*)d_in[1];   // (B,KG,H)
  const float* enc_h  = (const float*)d_in[2];   // (1,B,H)
  const float* W_a    = (const float*)d_in[3];   // (H,H)
  const float* emb    = (const float*)d_in[4];   // (V,H)
  const float* W_ih   = (const float*)d_in[5];   // (3H,H)
  const float* W_hh   = (const float*)d_in[6];   // (3H,H)
  const float* b_ih   = (const float*)d_in[7];   // (3H)
  const float* b_hh   = (const float*)d_in[8];   // (3H)
  const float* W_out  = (const float*)d_in[9];   // (V,3H)
  const float* b_out  = (const float*)d_in[10];  // (V)
  float* out = (float*)d_out;                    // (T,B,V)

  // --- carve workspace -----------------------------------------------------
  char* wp = (char*)d_ws;
  auto carve = [&](size_t bytes) -> void* {
    void* p = (void*)wp;
    wp += (bytes + 255) & ~(size_t)255;
    return p;
  };
  u16*   woutB = (u16*)carve((size_t)V * H3 * 2);     // 98.3 MB
  u16*   sentB = (u16*)carve((size_t)B * KS * H * 2); // 67.1 MB
  u16*   hasB  = (u16*)carve((size_t)B * KS * H * 2); // 67.1 MB
  u16*   grB   = (u16*)carve((size_t)B * KG * H * 2); // 16.8 MB
  u16*   hagB  = (u16*)carve((size_t)B * KG * H * 2); // 16.8 MB
  u16*   waT   = (u16*)carve((size_t)H * H * 2);
  float* scS   = (float*)carve((size_t)B * KS * 4);
  float* scG   = (float*)carve((size_t)B * KG * 4);
  float* hbuf0 = (float*)carve((size_t)B * H * 4);
  float* hbuf1 = (float*)carve((size_t)B * H * 4);
  u16*   X     = (u16*)carve((size_t)B * H3 * 2);
  int*   inp   = (int*)carve((size_t)B * 4);

  // --- one-time precompute -------------------------------------------------
  {
    size_t n = (size_t)V * H3;
    to_bf16_kernel<<<(unsigned)((n + 255) / 256), 256, 0, stream>>>(W_out, woutB, n);
    n = (size_t)B * KS * H;
    to_bf16_kernel<<<(unsigned)((n + 255) / 256), 256, 0, stream>>>(sent, sentB, n);
    n = (size_t)B * KG * H;
    to_bf16_kernel<<<(unsigned)((n + 255) / 256), 256, 0, stream>>>(graph, grB, n);
    transpose_bf16_kernel<<<(H * H + 255) / 256, 256, 0, stream>>>(W_a, waT, H, H);
  }
  // ha = enc @ W_a  (bf16 WMMA GEMMs, one-time ~43 GFLOP; MREP=2 reuses W_aT)
  {
    int M = B * KS, N = H, K = H;                 // 65536 x 512 x 512
    int waves = ((M >> 4) / 2) * (N >> 4);
    wmma_gemm_bt<false, false, 2><<<(waves + 7) / 8, 256, 0, stream>>>(
        sentB, waT, nullptr, hasB, M, N, K, N);
    M = B * KG;
    waves = ((M >> 4) / 2) * (N >> 4);
    wmma_gemm_bt<false, false, 2><<<(waves + 7) / 8, 256, 0, stream>>>(
        grB, waT, nullptr, hagB, M, N, K, N);
  }
  // h0 = encoder_hidden[0]; inp0 = BOS
  hipMemcpyAsync(hbuf0, enc_h, (size_t)B * H * 4, hipMemcpyDeviceToDevice, stream);
  fill_int_kernel<<<1, 32, 0, stream>>>(inp, 1 /*BOS*/, B);

  // --- sequential decode ---------------------------------------------------
  float* hcur = hbuf0;
  float* hnext = hbuf1;
  for (int t = 0; t < T; ++t) {
    // attention scores (wave-per-key dot products against hcur)
    attn_score_kernel<true ><<<dim3(KS / 8, B), 256, 0, stream>>>(hasB, hcur, scS, KS);
    attn_score_kernel<false><<<dim3(KG / 8, B), 256, 0, stream>>>(hagB, hcur, scG, KG);
    softmax_row_kernel<<<B, 256, 0, stream>>>(scS, KS);
    softmax_row_kernel<<<B, 256, 0, stream>>>(scG, KG);
    // contexts into X[:, 512:1024]=ct_g, X[:, 1024:1536]=ct_s
    attn_context_kernel<false><<<dim3(H / 256, B), 256, 0, stream>>>(grB,   scG, X, KG, 512);
    attn_context_kernel<true ><<<dim3(H / 256, B), 256, 0, stream>>>(sentB, scS, X, KS, 1024);
    // GRU: hnext = GRUCell(emb[inp], hcur); also X[:, 0:512] = hnext (bf16)
    gru_step_kernel<<<B, 512, 0, stream>>>(emb, inp, hcur, W_ih, W_hh, b_ih, b_hh,
                                           hnext, X);
    // out[t] = X @ W_out^T + b_out  (bf16 WMMA; one wave does both m-tiles
    // per n-tile -> each W_out fragment fetched once, 2 WMMAs per fetch)
    {
      int waves = ((B >> 4) / 2) * (V >> 4);   // 2000
      wmma_gemm_bt<true, true, 2><<<(waves + 7) / 8, 256, 0, stream>>>(
          X, woutB, b_out, out + (size_t)t * B * V, B, V, H3, V);
    }
    // next token
    argmax_row_kernel<<<B, 256, 0, stream>>>(out + (size_t)t * B * V, inp, V);
    // swap hidden buffers
    float* tmp = hcur; hcur = hnext; hnext = tmp;
  }
}
